// AttentionModule_16011638080155
// MI455X (gfx1250) — matrix operand order repaired and checked by reading, not compiled, not run on hardware
//
#include <hip/hip_runtime.h>

typedef _Float16 f16_t;
typedef __attribute__((ext_vector_type(8)))  _Float16 v8h;
typedef __attribute__((ext_vector_type(16))) _Float16 v16h;
typedef __attribute__((ext_vector_type(8)))  float    v8f;
typedef __attribute__((ext_vector_type(4)))  unsigned int u32x4;
typedef __attribute__((ext_vector_type(4)))  int i32x4;

union V16H { v16h v; v8h h[2]; };

#define BDIM 256
#define T_LEN 2048
#define L_LEN 2048
#define DMODEL 512
#define ADIM 512
#define NHEAD 8
#define HD 64

// ---------------------------------------------------------------------------
// Direct global -> LDS async copy (CDNA5 GLOBAL_LOAD_ASYNC_TO_LDS_B128,
// tracked by ASYNCcnt). Builtin expects int4 pointers: global (AS1) source,
// LDS (AS3) destination (per hipcc diagnostic on gfx1250).
// ---------------------------------------------------------------------------
#if defined(__has_builtin)
#if __has_builtin(__builtin_amdgcn_global_load_async_to_lds_b128)
#define HAS_ASYNC_LDS 1
#endif
#endif

#ifdef HAS_ASYNC_LDS
typedef __attribute__((address_space(1))) i32x4 gas_i32x4;
typedef __attribute__((address_space(3))) i32x4 las_i32x4;
#endif

__device__ __forceinline__ void copy16_g2l(f16_t* ldst, const f16_t* gsrc) {
#ifdef HAS_ASYNC_LDS
  __builtin_amdgcn_global_load_async_to_lds_b128((gas_i32x4*)gsrc,
                                                 (las_i32x4*)ldst, 0, 0);
#else
  *(u32x4*)ldst = *(const u32x4*)gsrc;
#endif
}

__device__ __forceinline__ void copy_wait() {
#ifdef HAS_ASYNC_LDS
#if defined(__has_builtin) && __has_builtin(__builtin_amdgcn_s_wait_asynccnt)
  __builtin_amdgcn_s_wait_asynccnt(0);
#else
  asm volatile("s_wait_asynccnt 0" ::: "memory");
#endif
#endif
}

// ---------------------------------------------------------------------------
// elementwise f32 -> f16
// ---------------------------------------------------------------------------
__global__ void f32_to_f16_kernel(const float* __restrict__ in,
                                  f16_t* __restrict__ out, long n) {
  long i = (long)blockIdx.x * blockDim.x + threadIdx.x;
  if (i < n) out[i] = (f16_t)in[i];
}

// ---------------------------------------------------------------------------
// x (B, 512, 2048) f32  ->  xT (B, 2048, 512) f16  (32x32 LDS tiles)
// ---------------------------------------------------------------------------
__global__ __launch_bounds__(BDIM) void transpose_x_kernel(
    const float* __restrict__ x, f16_t* __restrict__ xT) {
  __shared__ float tile[32][33];
  const int b = blockIdx.z;
  const int t0 = blockIdx.x * 32, d0 = blockIdx.y * 32;
  const int tx = threadIdx.x & 31, ty = threadIdx.x >> 5;  // ty 0..7
#pragma unroll
  for (int i = 0; i < 4; i++) {
    int d = ty + i * 8;
    tile[d][tx] = x[((size_t)b * DMODEL + d0 + d) * T_LEN + t0 + tx];
  }
  __syncthreads();
#pragma unroll
  for (int i = 0; i < 4; i++) {
    int t = ty + i * 8;
    xT[((size_t)b * T_LEN + t0 + t) * DMODEL + d0 + tx] = (f16_t)tile[tx][t];
  }
}

// ---------------------------------------------------------------------------
// len_q[b] = sum_t x_mask[b,0,t]; len_k[b] = sum_l context_mask[b,0,l]
// lens[0..3] = len_q, lens[4..7] = len_k
// ---------------------------------------------------------------------------
__global__ void lens_kernel(const float* __restrict__ xmask,
                            const float* __restrict__ cmask,
                            float* __restrict__ lens) {
  __shared__ float red[BDIM];
  const int bid = blockIdx.x;  // 0..7
  const float* src = (bid < 4) ? (xmask + (size_t)bid * T_LEN)
                               : (cmask + (size_t)(bid - 4) * L_LEN);
  float s = 0.f;
  for (int i = threadIdx.x; i < T_LEN; i += BDIM) s += src[i];
  red[threadIdx.x] = s;
  __syncthreads();
  for (int st = BDIM / 2; st > 0; st >>= 1) {
    if (threadIdx.x < st) red[threadIdx.x] += red[threadIdx.x + st];
    __syncthreads();
  }
  if (threadIdx.x == 0) lens[bid] = red[0];
}

// ---------------------------------------------------------------------------
// GEMM mainloop: C(128x64) = A(128x512) * W(n= nbase..nbase+64, k=512)^T
// 8 waves, each 32x32 (2x2 WMMA tiles), BK = 64, f16 in / f32 acc.
// Result left in Clds (stride 68 floats). Caller does epilogue.
// ---------------------------------------------------------------------------
__device__ __forceinline__ void gemm_tile_128x64(
    const f16_t* __restrict__ A, const f16_t* __restrict__ W, int m0, int nbase,
    f16_t* Alds, f16_t* Wlds, float* Clds, int tid) {
  const int w = tid >> 5, lane = tid & 31, l15 = lane & 15;
  const bool hi = lane >= 16;
  const int hi8 = hi ? 8 : 0;
  const int wm = (w >> 1) * 32, wn = (w & 1) * 32;
  v8f acc[2][2] = {};

  for (int k0 = 0; k0 < 512; k0 += 64) {
    // stage A: 128 rows x 64 halves (stride 72)
#pragma unroll
    for (int i = 0; i < 4; i++) {
      int idx = i * BDIM + tid;
      int r = idx >> 3, c = (idx & 7) * 8;
      copy16_g2l(Alds + r * 72 + c, A + (size_t)(m0 + r) * 512 + k0 + c);
    }
    // stage W: 64 rows (n) x 64 halves (stride 72)
#pragma unroll
    for (int i = 0; i < 2; i++) {
      int idx = i * BDIM + tid;
      int r = idx >> 3, c = (idx & 7) * 8;
      copy16_g2l(Wlds + r * 72 + c, W + (size_t)(nbase + r) * 512 + k0 + c);
    }
    copy_wait();
    __syncthreads();
#pragma unroll
    for (int kk = 0; kk < 64; kk += 32) {
      V16H a[2], bb[2];
#pragma unroll
      for (int sm = 0; sm < 2; sm++) {
        const f16_t* ap = Alds + (wm + sm * 16 + l15) * 72;
        a[sm].h[0] = *(const v8h*)(ap + kk + (hi ? 16 : 0));
        a[sm].h[1] = *(const v8h*)(ap + kk + 8 + (hi ? 16 : 0));
      }
#pragma unroll
      for (int sn = 0; sn < 2; sn++) {
        const f16_t* bp = Wlds + (wn + sn * 16 + l15) * 72 + kk + (hi ? 16 : 0);
        bb[sn].h[0] = *(const v8h*)bp;
        bb[sn].h[1] = *(const v8h*)(bp + 8);
      }
#pragma unroll
      for (int sm = 0; sm < 2; sm++)
#pragma unroll
        for (int sn = 0; sn < 2; sn++)
          acc[sm][sn] = __builtin_amdgcn_wmma_f32_16x16x32_f16(
              false, a[sm].v, false, bb[sn].v, (short)0, acc[sm][sn], false,
              false);
    }
    __syncthreads();
  }
#pragma unroll
  for (int sm = 0; sm < 2; sm++)
#pragma unroll
    for (int sn = 0; sn < 2; sn++)
#pragma unroll
      for (int r = 0; r < 8; r++)
        Clds[(wm + sm * 16 + r + hi8) * 68 + wn + sn * 16 + l15] =
            acc[sm][sn][r];
  __syncthreads();
}

// ---------------------------------------------------------------------------
// QKV projection. mode 0: RoPE epilogue -> out[b][h][m][64] (q or k)
//                 mode 1: plain epilogue, transposed -> out[b][h][hd][m] (v)
// blockIdx: (m-tile, head, batch)
// ---------------------------------------------------------------------------
__global__ __launch_bounds__(BDIM) void proj_gemm_kernel(
    const f16_t* __restrict__ A, const f16_t* __restrict__ W,
    const float* __restrict__ bias, f16_t* __restrict__ out,
    const float* __restrict__ lens, int M, int mode) {
  const int b = blockIdx.z, h = blockIdx.y, m0 = blockIdx.x * 128;
  const int tid = threadIdx.x;
  extern __shared__ char smem_raw[];
  f16_t* Alds = (f16_t*)smem_raw;          // 128 x 72
  f16_t* Wlds = Alds + 128 * 72;           // 64 x 72
  float* Clds = (float*)(Wlds + 64 * 72);  // 128 x 68

  gemm_tile_128x64(A + (size_t)b * M * 512, W, m0, h * 64, Alds, Wlds, Clds,
                   tid);

  if (mode == 0) {
    const float inv_len = 1.0f / lens[b];
    for (int idx = tid; idx < 128 * 32; idx += BDIM) {
      int row = idx & 127, i = idx >> 7;  // i in 0..31 (rotation pair index)
      float c1 = Clds[row * 68 + i] + bias[h * 64 + i];
      float c2 = Clds[row * 68 + i + 32] + bias[h * 64 + i + 32];
      float pos = (float)(m0 + row);
      // theta_i = 10 * 10000^(-i/32) = 10 * exp(-i * ln(10000)/32)
      float theta = 10.0f * __expf(-(float)i * 0.28782313662425572f);
      float f = pos * inv_len * theta;
      float sn, cs;
      __sincosf(f, &sn, &cs);
      size_t base = ((size_t)(b * NHEAD + h) * M + m0 + row) * 64;
      out[base + i] = (f16_t)(c1 * cs - c2 * sn);
      out[base + i + 32] = (f16_t)(c1 * sn + c2 * cs);
    }
  } else {
    for (int idx = tid; idx < 64 * 128; idx += BDIM) {
      int l = idx & 127, d = idx >> 7;  // d in 0..63
      float cv = Clds[l * 68 + d] + bias[h * 64 + d];
      out[((size_t)(b * NHEAD + h) * 64 + d) * M + m0 + l] = (f16_t)cv;
    }
  }
}

// ---------------------------------------------------------------------------
// Output projection: attn16 (B,T,512) @ Wo^T + bo, * x_mask, store (B,512,T)
// ---------------------------------------------------------------------------
__global__ __launch_bounds__(BDIM) void out_proj_kernel(
    const f16_t* __restrict__ A, const f16_t* __restrict__ W,
    const float* __restrict__ bias, const float* __restrict__ xmask,
    float* __restrict__ out, int M) {
  const int b = blockIdx.z, nb = blockIdx.y * 64, m0 = blockIdx.x * 128;
  const int tid = threadIdx.x;
  extern __shared__ char smem_raw[];
  f16_t* Alds = (f16_t*)smem_raw;
  f16_t* Wlds = Alds + 128 * 72;
  float* Clds = (float*)(Wlds + 64 * 72);

  gemm_tile_128x64(A + (size_t)b * M * 512, W, m0, nb, Alds, Wlds, Clds, tid);

  for (int idx = tid; idx < 64 * 128; idx += BDIM) {
    int t = idx & 127, n = idx >> 7;
    float cv = Clds[t * 68 + n] + bias[nb + n];
    cv *= xmask[(size_t)b * M + m0 + t];
    out[((size_t)b * DMODEL + nb + n) * M + m0 + t] = cv;
  }
}

// ---------------------------------------------------------------------------
// Flash attention per (b, h, 64-query tile), streaming 64-key chunks.
// q16:  [b][h][t][64]    k16: [b][h][l][64]    v16T: [b][h][hd][l]
// attn16 out: [b][t][h*64+hd]
// ---------------------------------------------------------------------------
__global__ __launch_bounds__(BDIM) void attn_kernel(
    const f16_t* __restrict__ q16, const f16_t* __restrict__ k16,
    const f16_t* __restrict__ v16T, const float* __restrict__ cmask,
    const float* __restrict__ xmask, f16_t* __restrict__ attn16) {
  const int b = blockIdx.z, h = blockIdx.y, t0 = blockIdx.x * 64;
  const int tid = threadIdx.x, w = tid >> 5, lane = tid & 31, l15 = lane & 15;
  const bool hi = lane >= 16;
  const int hi8 = hi ? 8 : 0;
  extern __shared__ char smem_raw[];
  f16_t* Qs = (f16_t*)smem_raw;        // 64 x 72
  f16_t* Ks = Qs + 64 * 72;            // 64 x 72
  f16_t* Vs = Ks + 64 * 72;            // 64(hd) x 72(l-chunk)
  f16_t* Ps = Vs + 64 * 72;            // 64 x 72
  float* Ss = (float*)(Ps + 64 * 72);  // 64 x 68
  float* mst = Ss + 64 * 68;           // 64
  float* lst = mst + 64;               // 64
  float* fac = lst + 64;               // 64
  float* red = fac + 64;               // 64 x 4

  const f16_t* Qg = q16 + ((size_t)(b * NHEAD + h) * T_LEN + t0) * HD;
  const f16_t* Kg = k16 + (size_t)(b * NHEAD + h) * L_LEN * HD;
  const f16_t* Vg = v16T + (size_t)(b * NHEAD + h) * HD * L_LEN;

  // stage Q tile (64 x 64 halves)
#pragma unroll
  for (int i = 0; i < 2; i++) {
    int idx = i * BDIM + tid;
    int r = idx >> 3, c = (idx & 7) * 8;
    copy16_g2l(Qs + r * 72 + c, Qg + (size_t)r * HD + c);
  }
  if (tid < 64) {
    mst[tid] = -1e30f;
    lst[tid] = 0.f;
  }
  v8f oacc[2] = {};
  const int wm = (w >> 1) * 16;   // wave m-tile (both S and PV)
  const int sn_b = (w & 1) * 32;  // S wave n base (2 subtiles)
  const int pvn = (w & 1) * 32;   // PV wave hd base (2 subtiles)
  const float inv_scale = 0.04419417382415922f;  // 1/sqrt(512)
  const int row = tid >> 2, sub = tid & 3;       // softmax thread mapping
  copy_wait();
  __syncthreads();

  for (int l0 = 0; l0 < L_LEN; l0 += 64) {
    // stage K chunk (64 rows x 64 halves) and V chunk (64 hd-rows x 64 halves)
#pragma unroll
    for (int i = 0; i < 2; i++) {
      int idx = i * BDIM + tid;
      int r = idx >> 3, c = (idx & 7) * 8;
      copy16_g2l(Ks + r * 72 + c, Kg + (size_t)(l0 + r) * HD + c);
    }
#pragma unroll
    for (int i = 0; i < 2; i++) {
      int idx = i * BDIM + tid;
      int r = idx >> 3, c = (idx & 7) * 8;
      copy16_g2l(Vs + r * 72 + c, Vg + (size_t)r * L_LEN + l0 + c);
    }
    // pull next chunk toward this WGP while we compute (L2-resident data)
    if (l0 + 64 < L_LEN) {
      __builtin_prefetch(Kg + (size_t)(l0 + 64 + (tid & 63)) * HD, 0, 0);
      __builtin_prefetch(Vg + (size_t)(tid & 63) * L_LEN + l0 + 64, 0, 0);
    }
    copy_wait();
    __syncthreads();

    // S = Q @ K^T : wave -> rows wm..wm+16, cols sn_b..sn_b+32
    v8f sacc[2] = {};
#pragma unroll
    for (int kk = 0; kk < 64; kk += 32) {
      V16H a;
      const f16_t* ap = Qs + (wm + l15) * 72;
      a.h[0] = *(const v8h*)(ap + kk + (hi ? 16 : 0));
      a.h[1] = *(const v8h*)(ap + kk + 8 + (hi ? 16 : 0));
#pragma unroll
      for (int sn = 0; sn < 2; sn++) {
        V16H bb;
        const f16_t* bp = Ks + (sn_b + sn * 16 + l15) * 72 + kk + (hi ? 16 : 0);
        bb.h[0] = *(const v8h*)bp;
        bb.h[1] = *(const v8h*)(bp + 8);
        sacc[sn] = __builtin_amdgcn_wmma_f32_16x16x32_f16(
            false, a.v, false, bb.v, (short)0, sacc[sn], false, false);
      }
    }
#pragma unroll
    for (int sn = 0; sn < 2; sn++)
#pragma unroll
      for (int r = 0; r < 8; r++)
        Ss[(wm + r + hi8) * 68 + sn_b + sn * 16 + l15] = sacc[sn][r];
    __syncthreads();

    // softmax pass1: scale + mask + partial row max (4 threads/row, 16 cols ea)
    float mloc = -1e30f;
#pragma unroll
    for (int c = 0; c < 16; c++) {
      int col = sub * 16 + c;
      float s = Ss[row * 68 + col] * inv_scale;
      if (cmask[(size_t)b * L_LEN + l0 + col] == 0.f) s = -1e30f;
      Ss[row * 68 + col] = s;
      mloc = fmaxf(mloc, s);
    }
    red[row * 4 + sub] = mloc;
    __syncthreads();
    if (tid < 64) {
      float rm = fmaxf(fmaxf(red[tid * 4 + 0], red[tid * 4 + 1]),
                       fmaxf(red[tid * 4 + 2], red[tid * 4 + 3]));
      float mold = mst[tid];
      float mnew = fmaxf(mold, rm);
      mst[tid] = mnew;
      fac[tid] = __expf(mold - mnew);
    }
    __syncthreads();
    // pass3: P = exp(S - m), partial sums, and rescale O accumulators
    {
      float mnew = mst[row];
      float ssum = 0.f;
#pragma unroll
      for (int c = 0; c < 16; c++) {
        int col = sub * 16 + c;
        float p = __expf(Ss[row * 68 + col] - mnew);
        ssum += p;
        Ps[row * 72 + col] = (f16_t)p;
      }
      red[row * 4 + sub] = ssum;
    }
#pragma unroll
    for (int t2 = 0; t2 < 2; t2++)
#pragma unroll
      for (int r = 0; r < 8; r++) oacc[t2][r] *= fac[wm + r + hi8];
    __syncthreads();
    if (tid < 64)
      lst[tid] = lst[tid] * fac[tid] + (red[tid * 4 + 0] + red[tid * 4 + 1] +
                                        red[tid * 4 + 2] + red[tid * 4 + 3]);

    // O += P @ V : wave -> rows wm..wm+16, hd cols pvn..pvn+32
#pragma unroll
    for (int kk = 0; kk < 64; kk += 32) {
      V16H a;
      const f16_t* ap = Ps + (wm + l15) * 72;
      a.h[0] = *(const v8h*)(ap + kk + (hi ? 16 : 0));
      a.h[1] = *(const v8h*)(ap + kk + 8 + (hi ? 16 : 0));
#pragma unroll
      for (int t2 = 0; t2 < 2; t2++) {
        V16H bb;
        const f16_t* bp = Vs + (pvn + t2 * 16 + l15) * 72 + kk + (hi ? 16 : 0);
        bb.h[0] = *(const v8h*)bp;
        bb.h[1] = *(const v8h*)(bp + 8);
        oacc[t2] = __builtin_amdgcn_wmma_f32_16x16x32_f16(
            false, a.v, false, bb.v, (short)0, oacc[t2], false, false);
      }
    }
    __syncthreads();
  }

  // finalize: O /= l, zero masked query rows, write f16
  if (tid < 64) {
    float l = lst[tid];
    float inv = (l > 0.f) ? 1.0f / l : 0.f;
    if (xmask[(size_t)b * T_LEN + t0 + tid] == 0.f) inv = 0.f;
    fac[tid] = inv;
  }
  __syncthreads();
#pragma unroll
  for (int t2 = 0; t2 < 2; t2++)
#pragma unroll
    for (int r = 0; r < 8; r++) {
      int rr = wm + r + hi8, col = pvn + t2 * 16 + l15;
      float o = oacc[t2][r] * fac[rr];
      attn16[((size_t)b * T_LEN + t0 + rr) * ADIM + h * 64 + col] = (f16_t)o;
    }
}

// ---------------------------------------------------------------------------
extern "C" void kernel_launch(void* const* d_in, const int* in_sizes, int n_in,
                              void* d_out, int out_size, void* d_ws,
                              size_t ws_size, hipStream_t stream) {
  (void)in_sizes; (void)n_in; (void)out_size; (void)ws_size;
  const float* x = (const float*)d_in[0];
  const float* context = (const float*)d_in[1];
  const float* x_mask = (const float*)d_in[2];
  const float* context_mask = (const float*)d_in[3];
  const float* Wq = (const float*)d_in[4];
  const float* bq = (const float*)d_in[5];
  const float* Wk = (const float*)d_in[6];
  const float* bk = (const float*)d_in[7];
  const float* Wv = (const float*)d_in[8];
  const float* bv = (const float*)d_in[9];
  const float* Wo = (const float*)d_in[10];
  const float* bo = (const float*)d_in[11];
  float* out = (float*)d_out;

  char* ws = (char*)d_ws;
  size_t off = 0;
  auto alloc = [&](size_t bytes) {
    void* p = ws + off;
    off += (bytes + 255) & ~(size_t)255;
    return p;
  };
  const size_t WELEM = 512 * 512;               // weight elements
  const size_t AELEM = (size_t)4 * 2048 * 512;  // activation elements
  f16_t* wq16 = (f16_t*)alloc(WELEM * 2);
  f16_t* wk16 = (f16_t*)alloc(WELEM * 2);
  f16_t* wv16 = (f16_t*)alloc(WELEM * 2);
  f16_t* wo16 = (f16_t*)alloc(WELEM * 2);
  f16_t* xT16 = (f16_t*)alloc(AELEM * 2);
  f16_t* ctx16 = (f16_t*)alloc(AELEM * 2);
  f16_t* q16 = (f16_t*)alloc(AELEM * 2);
  f16_t* k16 = (f16_t*)alloc(AELEM * 2);
  f16_t* v16T = (f16_t*)alloc(AELEM * 2);
  f16_t* attn16 = (f16_t*)alloc(AELEM * 2);
  float* lens = (float*)alloc(8 * sizeof(float));

  // 1) weight + context conversion to f16
  f32_to_f16_kernel<<<(int)(WELEM / BDIM), BDIM, 0, stream>>>(Wq, wq16, WELEM);
  f32_to_f16_kernel<<<(int)(WELEM / BDIM), BDIM, 0, stream>>>(Wk, wk16, WELEM);
  f32_to_f16_kernel<<<(int)(WELEM / BDIM), BDIM, 0, stream>>>(Wv, wv16, WELEM);
  f32_to_f16_kernel<<<(int)(WELEM / BDIM), BDIM, 0, stream>>>(Wo, wo16, WELEM);
  f32_to_f16_kernel<<<(int)(AELEM / BDIM), BDIM, 0, stream>>>(context, ctx16,
                                                              (long)AELEM);
  // 2) transpose x -> (B, T, 512) f16
  transpose_x_kernel<<<dim3(2048 / 32, 512 / 32, 4), BDIM, 0, stream>>>(x,
                                                                        xT16);
  // 3) mask lengths
  lens_kernel<<<8, BDIM, 0, stream>>>(x_mask, context_mask, lens);

  // 4) QKV projections (WMMA GEMMs) with RoPE / transposed-V epilogues
  const size_t gemm_lds =
      (size_t)(128 * 72 + 64 * 72) * 2 + (size_t)128 * 68 * 4;  // ~62.5 KB
  dim3 pg(2048 / 128, NHEAD, 4);
  proj_gemm_kernel<<<pg, BDIM, gemm_lds, stream>>>(xT16, wq16, bq, q16, lens,
                                                   2048, 0);
  proj_gemm_kernel<<<pg, BDIM, gemm_lds, stream>>>(ctx16, wk16, bk, k16,
                                                   lens + 4, 2048, 0);
  proj_gemm_kernel<<<pg, BDIM, gemm_lds, stream>>>(ctx16, wv16, bv, v16T, lens,
                                                   2048, 1);

  // 5) flash attention (WMMA QK^T and PV, fp32 online softmax)
  const size_t attn_lds = (size_t)(4 * 64 * 72) * 2 +
                          (size_t)(64 * 68 + 64 * 3 + 64 * 4) * 4;  // ~55 KB
  attn_kernel<<<dim3(2048 / 64, NHEAD, 4), BDIM, attn_lds, stream>>>(
      q16, k16, v16T, context_mask, x_mask, attn16);

  // 6) output projection + x_mask + transposed fp32 store
  out_proj_kernel<<<dim3(2048 / 128, 512 / 64, 4), BDIM, gemm_lds, stream>>>(
      attn16, wo16, bo, x_mask, out, 2048);
}
